// ResNetAttnBlock_8504035246389
// MI455X (gfx1250) — matrix operand order repaired and checked by reading, not compiled, not run on hardware
//
#include <hip/hip_runtime.h>
#include <hip/hip_bf16.h>
#include <math.h>

// ---------------------------------------------------------------------------
// ResNetBlock + SelfAttention for MI455X (gfx1250), wave32 + WMMA bf16.
//
// B=4, C=256, H=W=64 (HW=4096). All dense math -> v_wmma_f32_16x16x32_bf16.
// All WMMA B-operands are contiguous along K:
//   * activations / q / k / attn-o are stored token-major [B][HW][C] (bf16)
//   * v is channel-major [B][C][HW] (PV contracts over t)
// Attention is flash-style (online softmax) - no HW x HW score tensor in HBM.
// Q tile is DMA'd into LDS with the Tensor Data Mover (tensor_load_to_lds +
// s_wait_tensorcnt) using the 6-arg builtin this toolchain declares.
// ---------------------------------------------------------------------------

typedef __bf16 bf16_t;
typedef __attribute__((ext_vector_type(16))) __bf16 v16bf;
typedef __attribute__((ext_vector_type(8)))  __bf16 v8bf;
typedef __attribute__((ext_vector_type(8)))  float  v8f;
typedef __attribute__((ext_vector_type(4)))  unsigned int v4u;
typedef __attribute__((ext_vector_type(8)))  int v8i;
typedef __attribute__((ext_vector_type(4)))  int v4i;

#define HW    4096
#define CCH   256
#define BATCH 4

// A-matrix (16x32 bf16), CDNA5 ISA 7.12.2: lane half 0 -> K {0..7,16..23},
// half 1 -> K {8..15,24..31}. Two contiguous 16B runs per lane.
__device__ __forceinline__ v16bf ld_a(const bf16_t* row, int half) {
    union { v16bf v; v8bf h[2]; } u;
    u.h[0] = *(const v8bf*)(row + half * 8);
    u.h[1] = *(const v8bf*)(row + 16 + half * 8);
    return u.v;
}
// B-matrix (32x16 bf16): same K order as A: half 0 -> K {0..7,16..23}, half 1 -> K {8..15,24..31}.
__device__ __forceinline__ v16bf ld_b(const bf16_t* p, int half) {
    union { v16bf v; v8bf h[2]; } u;
    u.h[0] = *(const v8bf*)(p + half * 8);
    u.h[1] = *(const v8bf*)(p + 16 + half * 8);
    return u.v;
}
__device__ __forceinline__ v16bf zero16() { v16bf z = {}; return z; }

// ---------------------------------------------------------------------------
// Weight prep: fp32 -> bf16; conv weights transposed to [tap][oc][ic].
// ---------------------------------------------------------------------------
__global__ __launch_bounds__(256)
void cvt_bf16_kernel(const float* __restrict__ in, bf16_t* __restrict__ out, int n) {
    int i = blockIdx.x * 256 + threadIdx.x;
    if (i < n) out[i] = (bf16_t)in[i];
}

__global__ __launch_bounds__(256)
void cvt_convw_kernel(const float* __restrict__ w, bf16_t* __restrict__ out) {
    int i = blockIdx.x * 256 + threadIdx.x;          // i = (oc*256+ic)*9 + tap
    if (i < 9 * CCH * CCH) {
        int tap = i % 9;
        int rem = i / 9;
        int ic = rem & 255;
        int oc = rem >> 8;
        out[((size_t)tap * CCH + oc) * CCH + ic] = (bf16_t)w[i];
    }
}

// ---------------------------------------------------------------------------
// FiLM MLP: gb[b, r] = silu(te[b,:]) . mlp_w[r,:] + mlp_b[r]   (4 x 512)
// ---------------------------------------------------------------------------
__global__ __launch_bounds__(256)
void film_kernel(const float* __restrict__ te, const float* __restrict__ mlp_w,
                 const float* __restrict__ mlp_b, float* __restrict__ gb) {
    int j = blockIdx.x * 256 + threadIdx.x;
    if (j >= BATCH * 512) return;
    int b = j >> 9, r = j & 511;
    const float* t = te + b * 256;
    const float* w = mlp_w + (size_t)r * 256;
    float acc = mlp_b[r];
    for (int k = 0; k < 256; ++k) {
        float s = t[k];
        s = s / (1.f + __expf(-s));
        acc += s * w[k];
    }
    gb[j] = acc;
}

// ---------------------------------------------------------------------------
// GroupNorm (32 groups x 8 ch), optional FiLM before stats, optional SiLU.
// Input fp32 channel-major; output bf16 TOKEN-major [B][HW][C] so it can be a
// WMMA B-operand with contiguous-K loads. One block per (group, batch).
// ---------------------------------------------------------------------------
template<bool SILU, bool FILM>
__global__ __launch_bounds__(256)
void gn_kernel(const float* __restrict__ x, const float* __restrict__ scale,
               const float* __restrict__ bias, const float* __restrict__ film,
               bf16_t* __restrict__ out) {
    const int CPG = 8;
    const int NEL = CPG * HW;
    int g = blockIdx.x, b = blockIdx.y;
    int tid = threadIdx.x;
    __shared__ float s_sum[256], s_sq[256];

    const size_t base = ((size_t)b * CCH + g * CPG) * HW;
    float sum = 0.f, sq = 0.f;
    for (int i = tid; i < NEL; i += 256) {
        int c = g * CPG + (i >> 12);
        float v = x[base + i];
        if (FILM) v = (1.f + film[b * 512 + c]) * v + film[b * 512 + 256 + c];
        sum += v; sq += v * v;
    }
    s_sum[tid] = sum; s_sq[tid] = sq;
    __syncthreads();
    for (int off = 128; off > 0; off >>= 1) {
        if (tid < off) { s_sum[tid] += s_sum[tid + off]; s_sq[tid] += s_sq[tid + off]; }
        __syncthreads();
    }
    float mean = s_sum[0] * (1.f / NEL);
    float var  = s_sq[0] * (1.f / NEL) - mean * mean;
    float rstd = rsqrtf(var + 1e-5f);

    float sc[8], bi[8], ga[8], be[8];
#pragma unroll
    for (int cl = 0; cl < 8; ++cl) {
        int c = g * CPG + cl;
        sc[cl] = scale[c]; bi[cl] = bias[c];
        if (FILM) { ga[cl] = film[b * 512 + c]; be[cl] = film[b * 512 + 256 + c]; }
    }
    // one 16B token-major store per pixel
    for (int s = tid; s < HW; s += 256) {
        union { v8bf v; bf16_t e[8]; } o;
#pragma unroll
        for (int cl = 0; cl < 8; ++cl) {
            float v = x[base + (size_t)cl * HW + s];
            if (FILM) v = (1.f + ga[cl]) * v + be[cl];
            v = (v - mean) * rstd * sc[cl] + bi[cl];
            if (SILU) v = v / (1.f + __expf(-v));
            o.e[cl] = (bf16_t)v;
        }
        *(v8bf*)(out + ((size_t)b * HW + s) * CCH + g * CPG) = o.v;
    }
}

// ---------------------------------------------------------------------------
// Implicit-GEMM conv / 1x1 projection via WMMA.
//   out[b][oc][s] = sum_{tap,ic} Wt[tap][oc][ic] * actT[b][s+off(tap)][ic]
// actT token-major -> B-tiles are two b128 loads per lane.
// One wave per 16(oc) x 16(s) tile.  OUT_TRANS writes bf16 token-major with a
// single 16B store per lane-half (oc run is contiguous in r).
// ---------------------------------------------------------------------------
template<int TAPS, bool OUT_BF16, bool ADD_RES, bool HAS_BIAS, bool OUT_TRANS>
__global__ __launch_bounds__(32)
void gemm_wmma_kernel(const bf16_t* __restrict__ Wt,   // [TAPS][OC][IC]
                      const bf16_t* __restrict__ actT, // [B][HW][IC]
                      const float*  __restrict__ bias, // [OC]
                      const float*  __restrict__ res,  // [B][OC][HW]
                      float*  __restrict__ outF,       // [B][OC][HW]
                      bf16_t* __restrict__ outB,       // trans: [B][HW][OC]
                      int OC, int IC) {
    int s0  = blockIdx.x * 16;
    int oc0 = blockIdx.y * 16;
    int b   = blockIdx.z;
    int lane = threadIdx.x;
    int m = lane & 15, half = lane >> 4;

    int s  = s0 + m;
    int hh = s >> 6, wp = s & 63;

    v8f acc = {};
    for (int tap = 0; tap < TAPS; ++tap) {
        int dy = (TAPS == 9) ? (tap / 3 - 1) : 0;
        int dx = (TAPS == 9) ? (tap % 3 - 1) : 0;
        int hs = hh + dy, wsx = wp + dx;
        bool valid = ((unsigned)hs < 64u) && ((unsigned)wsx < 64u);
        int sp = hs * 64 + wsx;

        const bf16_t* wbase = Wt + ((size_t)tap * OC + (oc0 + m)) * IC;
        const bf16_t* abase = actT + ((size_t)b * HW + sp) * IC;

        for (int icc = 0; icc < IC; icc += 32) {
            __builtin_prefetch(wbase + icc + 32, 0, 3);   // global_prefetch_b8
            v16bf av = ld_a(wbase + icc, half);
            v16bf bv = valid ? ld_b(abase + icc, half) : zero16();
            acc = __builtin_amdgcn_wmma_f32_16x16x32_bf16(
                      false, av, false, bv, (short)0, acc, false, false);
        }
    }
    // D layout: vgpr r -> oc = oc0 + r + 8*half, col s = s0 + (lane&15)
    if (OUT_TRANS) {
        union { v8bf v; bf16_t e[8]; } o;
#pragma unroll
        for (int r = 0; r < 8; ++r) {
            float v = acc[r];
            if (HAS_BIAS) v += bias[oc0 + r + 8 * half];
            o.e[r] = (bf16_t)v;
        }
        *(v8bf*)(outB + ((size_t)b * HW + s) * OC + oc0 + 8 * half) = o.v;
    } else {
#pragma unroll
        for (int r = 0; r < 8; ++r) {
            int oc = oc0 + r + 8 * half;
            float v = acc[r];
            if (HAS_BIAS) v += bias[oc];
            size_t idx = ((size_t)b * OC + oc) * HW + s;
            if (ADD_RES) v += res[idx];
            if (OUT_BF16) outB[idx] = (bf16_t)v;
            else          outF[idx] = v;
        }
    }
}

// ---------------------------------------------------------------------------
// Flash attention, scale 1/sqrt(256) = 1/16. One block (8 waves) per
// (16-query tile, batch). Q tile loaded into LDS by the Tensor Data Mover.
// qT/kT token-major [B][HW][C]; v channel-major [B][C][HW]; o token-major.
// ---------------------------------------------------------------------------
__global__ __launch_bounds__(256)
void attn_kernel(const bf16_t* __restrict__ qT,
                 const bf16_t* __restrict__ kT,
                 const bf16_t* __restrict__ vC,
                 bf16_t* __restrict__ oT) {
    const float sm_scale = 1.f / 16.f;
    int s0 = blockIdx.x * 16;
    int b  = blockIdx.y;
    int tid  = threadIdx.x;
    int lane = tid & 31, wave = tid >> 5;
    int m = lane & 15, half = lane >> 4;

    __shared__ bf16_t ldsQ[16][264];      // q tile  [s][c]  (8-elem row pad)
    __shared__ float  ldsS[16][516];      // logits  [s][t_local] (512 chunk)
    __shared__ bf16_t ldsP[16][520];      // probs   [s][t_local]
    __shared__ float  red[16][17];
    __shared__ float  rowM[16], rowZ[16], rowF[16];

    const bf16_t* qb = qT + (size_t)b * HW * CCH;
    const bf16_t* kb = kT + (size_t)b * HW * CCH;
    const bf16_t* vb0 = vC + (size_t)b * CCH * HW;

#if __has_builtin(__builtin_amdgcn_tensor_load_to_lds) && \
    __has_builtin(__builtin_amdgcn_s_wait_tensorcnt)
    // --- TDM: DMA 16 x 256 bf16 tile (row stride 512B) into padded LDS ---
    if (wave == 0) {
        unsigned lds_off = (unsigned)(size_t)(&ldsQ[0][0]);
        unsigned long long ga = (unsigned long long)(qb + (size_t)s0 * CCH);
        v4u g0;
        g0.x = 1u;                                   // count=1, user desc
        g0.y = lds_off;                              // lds_addr
        g0.z = (unsigned)(ga & 0xffffffffu);         // global_addr[31:0]
        g0.w = (unsigned)((ga >> 32) & 0x1ffffffu)   // global_addr[56:32]
             | (2u << 30);                           // type=2 (image)
        v8i g1;
        g1[0] = (1 << 16)                            // data_size = 2B
              | (1 << 20)                            // pad_enable
              | (6 << 22)                            // pad_interval: 128 DW
              | (3 << 25);                           // pad_amount:   4 DW
        g1[1] = (CCH & 0xffff) << 16;                // tensor_dim0 = 256
        g1[2] = (HW & 0xffff) << 16;                 // tensor_dim1 = 4096
        g1[3] = (CCH & 0xffff) << 16;                // tile_dim0 = 256
        g1[4] = 16;                                  // tile_dim1 = 16
        g1[5] = CCH;                                 // tensor_dim0_stride = 256
        g1[6] = 0;
        g1[7] = 0;
        v4i z4 = {};
        v8i z8 = {};
        __builtin_amdgcn_tensor_load_to_lds(g0, g1, z4, z4, z8, 0);
    }
    __builtin_amdgcn_s_wait_tensorcnt(0);            // TENSORcnt==0 (NOP for waves 1-7)
#else
    for (int idx = tid; idx < 512; idx += 256) {     // fallback: coop vector copy
        int r = idx >> 5, c8 = (idx & 31) * 8;
        *(v8bf*)&ldsQ[r][c8] = *(const v8bf*)(qb + ((size_t)(s0 + r)) * CCH + c8);
    }
#endif
    if (tid < 16) { rowM[tid] = -1e30f; rowZ[tid] = 0.f; }
    __syncthreads();

    v8f oacc0 = {}, oacc1 = {};

    for (int kc = 0; kc < 8; ++kc) {                 // 8 chunks x 512 keys
        int t_base = kc * 512;

        // ---- 1) scores: each wave computes 4 key tiles ----
        for (int j = 0; j < 4; ++j) {
            int tl0 = (wave * 4 + j) * 16;
            int kt  = t_base + tl0;
            v8f sacc = {};
            const bf16_t* krow = kb + ((size_t)(kt + m)) * CCH;   // B col n = m
            for (int icc = 0; icc < CCH; icc += 32) {
                v16bf av = ld_a(&ldsQ[m][icc], half);
                v16bf bv = ld_b(krow + icc, half);
                sacc = __builtin_amdgcn_wmma_f32_16x16x32_bf16(
                           false, av, false, bv, (short)0, sacc, false, false);
            }
#pragma unroll
            for (int r = 0; r < 8; ++r)
                ldsS[r + 8 * half][tl0 + m] = sacc[r] * sm_scale;
        }
        __syncthreads();

        // ---- 2) online softmax over the chunk ----
        {
            int row = tid >> 4, sub = tid & 15;
            float lm = -1e30f;
            for (int jj = sub; jj < 512; jj += 16) lm = fmaxf(lm, ldsS[row][jj]);
            red[row][sub] = lm;
            __syncthreads();
            if (sub == 0) {
                float cm = red[row][0];
                for (int q = 1; q < 16; ++q) cm = fmaxf(cm, red[row][q]);
                float nM = fmaxf(rowM[row], cm);
                rowF[row] = __expf(rowM[row] - nM);
                rowM[row] = nM;
            }
            __syncthreads();
            float nM = rowM[row];
            float ls = 0.f;
            for (int jj = sub; jj < 512; jj += 16) {
                float p = __expf(ldsS[row][jj] - nM);
                ldsP[row][jj] = (bf16_t)p;
                ls += p;
            }
            red[row][sub] = ls;
            __syncthreads();
            if (sub == 0) {
                float cz = 0.f;
                for (int q = 0; q < 16; ++q) cz += red[row][q];
                rowZ[row] = rowZ[row] * rowF[row] + cz;
            }
            __syncthreads();
        }

        // ---- 3) rescale O, accumulate P x V^T (2 c-tiles per wave) ----
#pragma unroll
        for (int r = 0; r < 8; ++r) {
            float f = rowF[r + 8 * half];
            oacc0[r] *= f;
            oacc1[r] *= f;
        }
#pragma unroll
        for (int ct = 0; ct < 2; ++ct) {
            int c0 = (wave * 2 + ct) * 16;
            v8f oa = (ct == 0) ? oacc0 : oacc1;
            const bf16_t* vrow = vb0 + ((size_t)(c0 + m)) * HW + t_base;
            for (int tt = 0; tt < 512; tt += 32) {
                v16bf av = ld_a(&ldsP[m][tt], half);
                v16bf bv = ld_b(vrow + tt, half);
                oa = __builtin_amdgcn_wmma_f32_16x16x32_bf16(
                         false, av, false, bv, (short)0, oa, false, false);
            }
            if (ct == 0) oacc0 = oa; else oacc1 = oa;
        }
        __syncthreads();
    }

    // ---- 4) normalize, write o token-major [s][c] ----
#pragma unroll
    for (int ct = 0; ct < 2; ++ct) {
        int c0 = (wave * 2 + ct) * 16;
        v8f oa = (ct == 0) ? oacc0 : oacc1;
#pragma unroll
        for (int r = 0; r < 8; ++r) {
            int srow = r + 8 * half;
            float v = oa[r] / rowZ[srow];
            oT[((size_t)b * HW + s0 + srow) * CCH + c0 + m] = (bf16_t)v;
        }
    }
}

// ---------------------------------------------------------------------------
// Launch
// ---------------------------------------------------------------------------
extern "C" void kernel_launch(void* const* d_in, const int* in_sizes, int n_in,
                              void* d_out, int out_size, void* d_ws, size_t ws_size,
                              hipStream_t stream) {
    const float* x       = (const float*)d_in[0];
    const float* te      = (const float*)d_in[1];
    const float* gn1_s   = (const float*)d_in[2];
    const float* gn1_b   = (const float*)d_in[3];
    const float* conv1_w = (const float*)d_in[4];
    const float* conv1_b = (const float*)d_in[5];
    const float* mlp_w   = (const float*)d_in[6];
    const float* mlp_b   = (const float*)d_in[7];
    const float* gn2_s   = (const float*)d_in[8];
    const float* gn2_b   = (const float*)d_in[9];
    const float* conv2_w = (const float*)d_in[10];
    const float* conv2_b = (const float*)d_in[11];
    const float* gnA_s   = (const float*)d_in[12];
    const float* gnA_b   = (const float*)d_in[13];
    const float* qkv_w   = (const float*)d_in[14];
    const float* out_w   = (const float*)d_in[15];
    const float* out_b   = (const float*)d_in[16];
    float* out = (float*)d_out;

    char* ws = (char*)d_ws;
    size_t off = 0;
    auto alloc = [&](size_t bytes) -> void* {
        void* p = ws + off;
        off = (off + bytes + 255) & ~((size_t)255);
        return p;
    };
    bf16_t* W1t  = (bf16_t*)alloc((size_t)9 * CCH * CCH * 2);
    bf16_t* W2t  = (bf16_t*)alloc((size_t)9 * CCH * CCH * 2);
    bf16_t* Wq   = (bf16_t*)alloc((size_t)3 * CCH * CCH * 2);
    bf16_t* Wo   = (bf16_t*)alloc((size_t)CCH * CCH * 2);
    float*  filmv= (float*) alloc((size_t)BATCH * 512 * 4);
    bf16_t* act1 = (bf16_t*)alloc((size_t)BATCH * HW * CCH * 2);  // token-major
    float*  hbuf = (float*) alloc((size_t)BATCH * CCH * HW * 4);
    bf16_t* act2 = (bf16_t*)alloc((size_t)BATCH * HW * CCH * 2);
    float*  xr   = (float*) alloc((size_t)BATCH * CCH * HW * 4);
    bf16_t* nrm  = (bf16_t*)alloc((size_t)BATCH * HW * CCH * 2);
    bf16_t* qTb  = (bf16_t*)alloc((size_t)BATCH * HW * CCH * 2);
    bf16_t* kTb  = (bf16_t*)alloc((size_t)BATCH * HW * CCH * 2);
    bf16_t* vCb  = (bf16_t*)alloc((size_t)BATCH * CCH * HW * 2);
    bf16_t* oTb  = (bf16_t*)alloc((size_t)BATCH * HW * CCH * 2);
    (void)ws_size; (void)in_sizes; (void)n_in; (void)out_size;

    // weight prep
    cvt_convw_kernel<<<(9 * CCH * CCH + 255) / 256, 256, 0, stream>>>(conv1_w, W1t);
    cvt_convw_kernel<<<(9 * CCH * CCH + 255) / 256, 256, 0, stream>>>(conv2_w, W2t);
    cvt_bf16_kernel<<<(3 * CCH * CCH + 255) / 256, 256, 0, stream>>>(qkv_w, Wq, 3 * CCH * CCH);
    cvt_bf16_kernel<<<(CCH * CCH + 255) / 256, 256, 0, stream>>>(out_w, Wo, CCH * CCH);

    // FiLM
    film_kernel<<<(BATCH * 512 + 255) / 256, 256, 0, stream>>>(te, mlp_w, mlp_b, filmv);

    // ResNet block
    gn_kernel<true, false><<<dim3(32, BATCH), 256, 0, stream>>>(x, gn1_s, gn1_b, nullptr, act1);
    gemm_wmma_kernel<9, false, false, true, false>
        <<<dim3(HW / 16, CCH / 16, BATCH), 32, 0, stream>>>(
            W1t, act1, conv1_b, nullptr, hbuf, nullptr, CCH, CCH);
    gn_kernel<true, true><<<dim3(32, BATCH), 256, 0, stream>>>(hbuf, gn2_s, gn2_b, filmv, act2);
    gemm_wmma_kernel<9, false, true, true, false>
        <<<dim3(HW / 16, CCH / 16, BATCH), 32, 0, stream>>>(
            W2t, act2, conv2_b, x, xr, nullptr, CCH, CCH);

    // Attention: GN -> q/k (token-major), v (channel-major)
    gn_kernel<false, false><<<dim3(32, BATCH), 256, 0, stream>>>(xr, gnA_s, gnA_b, nullptr, nrm);
    gemm_wmma_kernel<1, true, false, false, true>
        <<<dim3(HW / 16, CCH / 16, BATCH), 32, 0, stream>>>(
            Wq, nrm, nullptr, nullptr, nullptr, qTb, CCH, CCH);
    gemm_wmma_kernel<1, true, false, false, true>
        <<<dim3(HW / 16, CCH / 16, BATCH), 32, 0, stream>>>(
            Wq + (size_t)CCH * CCH, nrm, nullptr, nullptr, nullptr, kTb, CCH, CCH);
    gemm_wmma_kernel<1, true, false, false, false>
        <<<dim3(HW / 16, CCH / 16, BATCH), 32, 0, stream>>>(
            Wq + (size_t)2 * CCH * CCH, nrm, nullptr, nullptr, nullptr, vCb, CCH, CCH);

    attn_kernel<<<dim3(HW / 16, BATCH), 256, 0, stream>>>(qTb, kTb, vCb, oTb);

    gemm_wmma_kernel<1, false, true, true, false>
        <<<dim3(HW / 16, CCH / 16, BATCH), 32, 0, stream>>>(
            Wo, oTb, out_b, xr, out, nullptr, CCH, CCH);
}
